// PPO_23098334117920
// MI455X (gfx1250) — matrix operand order repaired and checked by reading, not compiled, not run on hardware
//
#include <hip/hip_runtime.h>
#include <hip/hip_bf16.h>
#include <math.h>

// ---------------------------------------------------------------------------
// ODE-gated LSTM recurrence on gfx1250 (MI455X).
// Persistent block per 16-row batch tile; bf16 weights + bf16 activation
// mirrors LDS-resident; all GEMMs via v_wmma_f32_16x16x32_bf16.
// One-hot(action) is folded into the [h,c,onehot] A-matrix so the xi GEMM is
// a pure 3-K-tile WMMA contraction (no per-element LDS gathers).
// Waves 0-1 run the serial ODE chain; waves 2-3 overlap the recurrence-
// independent x@Wih^T gates half, held in registers across barriers.
// ---------------------------------------------------------------------------

typedef __attribute__((ext_vector_type(16))) __bf16 v16bf;
typedef __attribute__((ext_vector_type(8)))  __bf16 v8bf;
typedef __attribute__((ext_vector_type(8)))  float  v8f;

#define WMMA_BF16(a, b, c) \
    __builtin_amdgcn_wmma_f32_16x16x32_bf16(false, (a), false, (b), (short)0, (c), false, false)

namespace {
constexpr int kB   = 2048;
constexpr int kT   = 256;
constexpr int kLD  = 32;    // latent dim
constexpr int kOBS = 7;     // 4 obs + 2 ts + 1 act

__device__ __forceinline__ float sigmoidf(float x) {
    return 1.0f / (1.0f + __expf(-x));
}

// A fragment (16xK tile, M rows) from a bf16 LDS buffer.
// ISA 7.12.2 lane layout: row = lane&15; elems 0..7 -> K = 8*(lane>>4)+e,
// elems 8..15 -> K = 16+8*(lane>>4)+e.  Two contiguous 16B chunks ->
// two ds_load_b128.
__device__ __forceinline__ v16bf load_Abf(const __bf16* act, int stride, int kt) {
    const int lane = threadIdx.x & 31;
    const __bf16* p = act + (lane & 15) * stride + kt * 32 + (lane >> 4) * 8;
    const v8bf lo = *(const v8bf*)(p);
    const v8bf hi = *(const v8bf*)(p + 16);
    return __builtin_shufflevector(lo, hi, 0, 1, 2, 3, 4, 5, 6, 7,
                                           8, 9, 10, 11, 12, 13, 14, 15);
}

// B fragment (32x16 KxN tile) from bf16 LDS weight stored as Wt[k][n]
// (row stride = Npad). Lane supplies N = nt*16+(lane&15); elems 0..7 -> K = 8*(lane>>4)+e,
// elems 8..15 -> K = 16+8*(lane>>4)+e (strided scalar loads).
__device__ __forceinline__ v16bf load_Bbf(const __bf16* wgt, int stride, int kt, int nt) {
    const int lane = threadIdx.x & 31;
    const __bf16* p = wgt + (kt * 32 + (lane >> 4) * 8) * stride + nt * 16 + (lane & 15);
    v16bf r;
#pragma unroll
    for (int e = 0; e < 8; ++e) {
        r[e]     = p[e * stride];
        r[8 + e] = p[(16 + e) * stride];
    }
    return r;
}

// Transposed bf16 weight upload: dst[k*Npad+n] = W[n*rowStride+k] (zero pad).
__device__ __forceinline__ void stage_weight(__bf16* dst, const float* W,
                                             int outDim, int inDim, int rowStride,
                                             int Kpad, int Npad, int tid, int nthr) {
    for (int idx = tid; idx < Kpad * Npad; idx += nthr) {
        const int k = idx / Npad;
        const int n = idx % Npad;
        float v = (k < inDim && n < outDim) ? W[n * rowStride + k] : 0.0f;
        dst[idx] = (__bf16)v;
    }
}

__device__ __forceinline__ void stage_bias(float* dst, const float* b,
                                           int n, int npad, int tid, int nthr) {
    for (int i = tid; i < npad; i += nthr) dst[i] = (i < n) ? b[i] : 0.0f;
}
} // namespace

__global__ __launch_bounds__(128)
void ode_lstm_persistent(const float* __restrict__ obs,
                         const float* __restrict__ h0,  const float* __restrict__ c0,
                         const float* __restrict__ W1,  const float* __restrict__ b1,
                         const float* __restrict__ Wih, const float* __restrict__ Whh,
                         const float* __restrict__ bih, const float* __restrict__ bhh,
                         const float* __restrict__ Wxi, const float* __restrict__ bxi,
                         const float* __restrict__ Wl1, const float* __restrict__ bl1,
                         const float* __restrict__ Wl2, const float* __restrict__ bl2,
                         const float* __restrict__ Wl3, const float* __restrict__ bl3,
                         float* __restrict__ out) {
    // ---- bf16 weights, K x N padded layouts (16B aligned for b128 loads) --
    __shared__ __align__(16) __bf16 sW1t [32 * 64];
    __shared__ __align__(16) __bf16 sWxit[96 * 64];   // K=96: [h,c]=0..63, onehot=64..66
    __shared__ __align__(16) __bf16 sWl1t[64 * 32];
    __shared__ __align__(16) __bf16 sWl2t[32 * 32];
    __shared__ __align__(16) __bf16 sWl3t[32 * 64];
    __shared__ __align__(16) __bf16 sWiht[64 * 128];
    __shared__ __align__(16) __bf16 sWhht[32 * 128];
    // ---- biases -----------------------------------------------------------
    __shared__ float sb1[64], sbxi[64], sbl1[32], sbl2[32], sbl3[64], sbg[128];
    // ---- bf16 activation mirrors (WMMA A operands) ------------------------
    __shared__ __align__(16) __bf16 sObAb[16 * 32];
    __shared__ __align__(16) __bf16 sHCb [16 * 96];   // [h | c | onehot | pad]
    __shared__ __align__(16) __bf16 sXb  [16 * 64];
    __shared__ __align__(16) __bf16 sZPb [16 * 64];
    __shared__ __align__(16) __bf16 sZ1b [16 * 32];
    __shared__ __align__(16) __bf16 sZ2b [16 * 32];
    __shared__ __align__(16) __bf16 sHinb[16 * 32];
    // ---- f32 state / elementwise buffers ----------------------------------
    __shared__ float sHC  [16 * 64];                  // [h | c] carried state (f32)
    __shared__ float sZP  [16 * 64];
    __shared__ float sHCin[16 * 64];
    __shared__ float sG   [16 * 128];
    __shared__ float sDT[16];
    __shared__ int   sCOND[16];

    const int tid  = threadIdx.x;
    const int w    = tid >> 5;                        // wave id 0..3
    const int lane = tid & 31;
    const int tile = blockIdx.x;
    const int nthr = blockDim.x;

    // ======================= one-time staging ==============================
    stage_weight(sW1t,  W1,  64,  4,  4, 32,  64, tid, nthr);
    stage_weight(sWxit, Wxi, 64, 67, 67, 96,  64, tid, nthr);  // rows 64..66 = act cols
    stage_weight(sWl1t, Wl1, 20, 64, 64, 64,  32, tid, nthr);
    stage_weight(sWl2t, Wl2, 20, 20, 20, 32,  32, tid, nthr);
    stage_weight(sWl3t, Wl3, 64, 20, 20, 32,  64, tid, nthr);
    stage_weight(sWiht, Wih, 128, 64, 64, 64, 128, tid, nthr);
    stage_weight(sWhht, Whh, 128, 32, 32, 32, 128, tid, nthr);
    stage_bias(sb1,  b1,  64, 64, tid, nthr);
    stage_bias(sbxi, bxi, 64, 64, tid, nthr);
    stage_bias(sbl1, bl1, 20, 32, tid, nthr);
    stage_bias(sbl2, bl2, 20, 32, tid, nthr);
    stage_bias(sbl3, bl3, 64, 64, tid, nthr);
    for (int i = tid; i < 128; i += nthr) sbg[i] = bih[i] + bhh[i];
    for (int i = tid; i < 16 * 96; i += nthr) {       // [h|c] + zero onehot/pad
        const int m = i / 96, j = i % 96;
        float v = 0.0f;
        if (j < kLD)            v = h0[(tile * 16 + m) * kLD + j];
        else if (j < 2 * kLD)   v = c0[(tile * 16 + m) * kLD + (j - kLD)];
        sHCb[i] = (__bf16)v;
        if (j < 2 * kLD) sHC[m * 64 + j] = v;
    }
    for (int i = tid; i < 16 * 32; i += nthr) sObAb[i] = (__bf16)0.0f;
    // preload obs for t = 0
    if (tid < 16) {
        const float* o = obs + ((size_t)(tile * 16 + tid) * kT + 0) * kOBS;
        sObAb[tid * 32 + 0] = (__bf16)o[0];
        sObAb[tid * 32 + 1] = (__bf16)o[1];
        sObAb[tid * 32 + 2] = (__bf16)o[2];
        sObAb[tid * 32 + 3] = (__bf16)o[3];
        sDT[tid]   = o[5] - o[4];
        sCOND[tid] = (o[4] + o[5]) != 0.0f;
        const int a = (int)o[6];
        sHCb[tid * 96 + 64] = (__bf16)(a == 0 ? 1.0f : 0.0f);
        sHCb[tid * 96 + 65] = (__bf16)(a == 1 ? 1.0f : 0.0f);
        sHCb[tid * 96 + 66] = (__bf16)(a == 2 ? 1.0f : 0.0f);
    }
    __syncthreads();

    // ======================= recurrence over time ==========================
#pragma unroll 1
    for (int t = 0; t < kT; ++t) {
        const int n16 = lane & 15;
        const int mb  = (lane >> 4) * 8;

        // ---- stage 1 (all waves): x = ob@W1^T+b1 ; zp = [h,c,1hot]@Wxi^T --
        {
            v8f accX = {};
            accX = WMMA_BF16(load_Abf(sObAb, 32, 0), load_Bbf(sW1t, 64, 0, w), accX);
            v8f accZ = {};
            accZ = WMMA_BF16(load_Abf(sHCb, 96, 0), load_Bbf(sWxit, 64, 0, w), accZ);
            accZ = WMMA_BF16(load_Abf(sHCb, 96, 1), load_Bbf(sWxit, 64, 1, w), accZ);
            accZ = WMMA_BF16(load_Abf(sHCb, 96, 2), load_Bbf(sWxit, 64, 2, w), accZ);
            const int n = n16 + w * 16;
#pragma unroll
            for (int v = 0; v < 8; ++v) {
                const int m = mb + v;
                sXb[m * 64 + n] = (__bf16)(accX[v] + sb1[n]);
                const float zpv = accZ[v] + sbxi[n];
                sZP [m * 64 + n] = zpv;
                sZPb[m * 64 + n] = (__bf16)zpv;
            }
        }
        __syncthreads();

        // ---- stage A: waves 0-1 ODE layer 1; waves 2-3 gates x-part -------
        v8f gx[4];
        if (w < 2) {
            v8f acc = {};
            acc = WMMA_BF16(load_Abf(sZPb, 64, 0), load_Bbf(sWl1t, 32, 0, w), acc);
            acc = WMMA_BF16(load_Abf(sZPb, 64, 1), load_Bbf(sWl1t, 32, 1, w), acc);
            const int n = n16 + w * 16;
#pragma unroll
            for (int v = 0; v < 8; ++v)
                sZ1b[(mb + v) * 32 + n] = (__bf16)fmaxf(acc[v] + sbl1[n], 0.0f);
        } else {
            const v16bf ax0 = load_Abf(sXb, 64, 0);   // shared across all 4 N-tiles
            const v16bf ax1 = load_Abf(sXb, 64, 1);
#pragma unroll
            for (int q = 0; q < 4; ++q) {
                const int nt = (w - 2) + 2 * q;       // wave2: 0,2,4,6  wave3: 1,3,5,7
                v8f acc = {};
                acc = WMMA_BF16(ax0, load_Bbf(sWiht, 128, 0, nt), acc);
                acc = WMMA_BF16(ax1, load_Bbf(sWiht, 128, 1, nt), acc);
                gx[q] = acc;                          // held in VGPRs across barriers
            }
        }
        __syncthreads();

        // ---- stage B: waves 0-1 ODE layer 2 -------------------------------
        if (w < 2) {
            v8f acc = {};
            acc = WMMA_BF16(load_Abf(sZ1b, 32, 0), load_Bbf(sWl2t, 32, 0, w), acc);
            const int n = n16 + w * 16;
#pragma unroll
            for (int v = 0; v < 8; ++v)
                sZ2b[(mb + v) * 32 + n] = (__bf16)fmaxf(acc[v] + sbl2[n], 0.0f);
        }
        __syncthreads();

        // ---- stage C: waves 0-1: fo = z2@Wl3^T+bl3, fused ODE jump+select -
        if (w < 2) {
            const v16bf az2 = load_Abf(sZ2b, 32, 0);
#pragma unroll
            for (int q = 0; q < 2; ++q) {
                const int nt = w + 2 * q;             // wave0: 0,2  wave1: 1,3
                v8f acc = {};
                acc = WMMA_BF16(az2, load_Bbf(sWl3t, 64, 0, nt), acc);
                const int n = n16 + nt * 16;
#pragma unroll
                for (int v = 0; v < 8; ++v) {
                    const int m = mb + v;
                    const float fov = acc[v] + sbl3[n];
                    const float z   = sZP[m * 64 + n] + sDT[m] * fov;
                    const float hv  = sCOND[m] ? z : sHC[m * 64 + n];
                    sHCin[m * 64 + n] = hv;
                    if (nt < 2)                       // h-half mirror for Whh GEMM
                        sHinb[m * 32 + n] = (__bf16)hv;
                }
            }
        }
        __syncthreads();

        // ---- stage D: waves 2-3: gates = gx + h_in@Whh^T + (bih+bhh) ------
        if (w >= 2) {
            const v16bf ah = load_Abf(sHinb, 32, 0);  // shared across all 4 N-tiles
#pragma unroll
            for (int q = 0; q < 4; ++q) {
                const int nt = (w - 2) + 2 * q;
                v8f acc = gx[q];
                acc = WMMA_BF16(ah, load_Bbf(sWhht, 128, 0, nt), acc);
                const int n = n16 + nt * 16;
#pragma unroll
                for (int v = 0; v < 8; ++v)
                    sG[(mb + v) * 128 + n] = acc[v] + sbg[n];
            }
        }
        __syncthreads();

        // ---- pointwise LSTM cell + output store + next-step obs preload ---
        if (tid < 16 && t + 1 < kT) {
            const float* o = obs + ((size_t)(tile * 16 + tid) * kT + (t + 1)) * kOBS;
            sObAb[tid * 32 + 0] = (__bf16)o[0];
            sObAb[tid * 32 + 1] = (__bf16)o[1];
            sObAb[tid * 32 + 2] = (__bf16)o[2];
            sObAb[tid * 32 + 3] = (__bf16)o[3];
            sDT[tid]   = o[5] - o[4];
            sCOND[tid] = (o[4] + o[5]) != 0.0f;
            const int a = (int)o[6];
            sHCb[tid * 96 + 64] = (__bf16)(a == 0 ? 1.0f : 0.0f);
            sHCb[tid * 96 + 65] = (__bf16)(a == 1 ? 1.0f : 0.0f);
            sHCb[tid * 96 + 66] = (__bf16)(a == 2 ? 1.0f : 0.0f);
            __builtin_prefetch(o + kOBS, 0, 3);       // global_prefetch_b8 (t+2)
        }
        for (int i = tid; i < 16 * kLD; i += nthr) {
            const int m = i >> 5, u = i & 31;
            const float ig = sG[m * 128 + u];
            const float fg = sG[m * 128 + 32 + u];
            const float gg = sG[m * 128 + 64 + u];
            const float og = sG[m * 128 + 96 + u];
            const float ci = sHCin[m * 64 + 32 + u];
            const float cn = sigmoidf(fg) * ci + sigmoidf(ig) * tanhf(gg);
            const float hn = sigmoidf(og) * tanhf(cn);
            sHC [m * 64 + u]      = hn;
            sHC [m * 64 + 32 + u] = cn;
            sHCb[m * 96 + u]      = (__bf16)hn;
            sHCb[m * 96 + 32 + u] = (__bf16)cn;
            out[((size_t)(tile * 16 + m) * kT + t) * kLD + u] = hn;
        }
        __syncthreads();
    }
}

extern "C" void kernel_launch(void* const* d_in, const int* in_sizes, int n_in,
                              void* d_out, int out_size, void* d_ws, size_t ws_size,
                              hipStream_t stream) {
    (void)in_sizes; (void)n_in; (void)out_size; (void)d_ws; (void)ws_size;
    const float* obs = (const float*)d_in[0];
    const float* h0  = (const float*)d_in[1];
    const float* c0  = (const float*)d_in[2];
    const float* W1  = (const float*)d_in[3];
    const float* b1  = (const float*)d_in[4];
    const float* Wih = (const float*)d_in[5];
    const float* Whh = (const float*)d_in[6];
    const float* bih = (const float*)d_in[7];
    const float* bhh = (const float*)d_in[8];
    const float* Wxi = (const float*)d_in[9];
    const float* bxi = (const float*)d_in[10];
    const float* Wl1 = (const float*)d_in[11];
    const float* bl1 = (const float*)d_in[12];
    const float* Wl2 = (const float*)d_in[13];
    const float* bl2 = (const float*)d_in[14];
    const float* Wl3 = (const float*)d_in[15];
    const float* bl3 = (const float*)d_in[16];
    float* out = (float*)d_out;

    dim3 grid(kB / 16);   // 128 persistent blocks, one 16-row tile each
    dim3 block(128);      // 4 wave32s
    ode_lstm_persistent<<<grid, block, 0, stream>>>(
        obs, h0, c0, W1, b1, Wih, Whh, bih, bhh, Wxi, bxi,
        Wl1, bl1, Wl2, bl2, Wl3, bl3, out);
}
